// GPT2Model_81698867905265
// MI455X (gfx1250) — matrix operand order repaired and checked by reading, not compiled, not run on hardware
//
#include <hip/hip_runtime.h>
#include <hip/hip_bf16.h>
#include <math.h>

// ---------------- problem dims (fixed by the reference) ----------------
#define B_  4
#define S_  2048
#define E_  256
#define H_  8
#define NI_ 1024
#define L_  2
#define M_  (B_ * S_)        // 8192 rows of activations

typedef unsigned short u16;
typedef __attribute__((ext_vector_type(4)))  unsigned int u32x4;
typedef __attribute__((ext_vector_type(16))) __bf16       v16bf;
typedef __attribute__((ext_vector_type(8)))  float        v8f;
typedef __attribute__((ext_vector_type(4)))  unsigned int uv4;
typedef __attribute__((ext_vector_type(8)))  int          iv8;
typedef __attribute__((ext_vector_type(4)))  int          iv4;

// 16 bf16 values per lane = 8 VGPRs; also viewable as two 16B chunks for b128 loads
union Frag {
    u32x4 q[2];
    v16bf v;
    u16   u[16];
};

// ---------------- small device helpers ----------------
__device__ __forceinline__ u16 f2bf(float f) {
    union { float f; unsigned int u; } c; c.f = f;
    unsigned int r = c.u + 0x7FFFu + ((c.u >> 16) & 1u);   // round-to-nearest-even
    return (u16)(r >> 16);
}
__device__ __forceinline__ float gelu_new(float x) {
    float x3 = x * x * x;
    return 0.5f * x * (1.0f + tanhf(0.7978845608028654f * (x + 0.044715f * x3)));
}
__device__ __forceinline__ float rmax16(float v) {
    v = fmaxf(v, __shfl_xor(v, 1, 32));
    v = fmaxf(v, __shfl_xor(v, 2, 32));
    v = fmaxf(v, __shfl_xor(v, 4, 32));
    v = fmaxf(v, __shfl_xor(v, 8, 32));
    return v;
}
__device__ __forceinline__ float rsum16(float v) {
    v += __shfl_xor(v, 1, 32);
    v += __shfl_xor(v, 2, 32);
    v += __shfl_xor(v, 4, 32);
    v += __shfl_xor(v, 8, 32);
    return v;
}
__device__ __forceinline__ float rsum32(float v) {
    v += __shfl_xor(v, 1, 32);
    v += __shfl_xor(v, 2, 32);
    v += __shfl_xor(v, 4, 32);
    v += __shfl_xor(v, 8, 32);
    v += __shfl_xor(v, 16, 32);
    return v;
}
__device__ __forceinline__ v8f wmma_bf16(const Frag& a, const Frag& b, v8f c) {
    return __builtin_amdgcn_wmma_f32_16x16x32_bf16(false, a.v, false, b.v,
                                                   (short)0, c, false, false);
}
// LDS byte offset (32-bit, from wave's LDS base) of a __shared__ object
__device__ __forceinline__ unsigned lds_offset(const void* p) {
    return (unsigned)(uintptr_t)(const __attribute__((address_space(3))) void*)p;
}

// Build a 2D-tile Tensor DMA Descriptor and issue TENSOR_LOAD_TO_LDS.
//   gbase:   global address of tile start (element granularity, bf16)
//   ldsaddr: destination LDS byte offset
//   td0/td1: tile dims (elements along contiguous dim0 / rows)
//   stride0: tensor dim0 stride in elements
//   dim0/dim1: tensor dims for OOB checking
__device__ __forceinline__ void tdm_load_2d(const u16* gbase, unsigned ldsaddr,
                                            unsigned td0, unsigned td1,
                                            unsigned stride0,
                                            unsigned dim0, unsigned dim1) {
    unsigned long long ga = (unsigned long long)(uintptr_t)gbase;
    uv4 g0;
    g0[0] = 1u;                                   // count=1 (valid), user descriptor
    g0[1] = ldsaddr;                              // lds_addr[31:0]
    g0[2] = (unsigned)(ga & 0xFFFFFFFFu);         // global_addr[31:0]
    g0[3] = (unsigned)((ga >> 32) & 0x01FFFFFFu)  // global_addr[56:32]
          | (2u << 30);                           // type = 2 ("image")
    unsigned long long s0 = (unsigned long long)stride0;
    iv8 g1;
    g1[0] = (int)(1u << 16);                      // workgroup_mask=0, data_size=1 (2B)
    g1[1] = (int)((dim0 & 0xFFFFu) << 16);        // tensor_dim0[15:0] (bits 79:48)
    g1[2] = (int)(((dim0 >> 16) & 0xFFFFu) | ((dim1 & 0xFFFFu) << 16));
    g1[3] = (int)(((dim1 >> 16) & 0xFFFFu) | ((td0 & 0xFFFFu) << 16)); // tile_dim0
    g1[4] = (int)(td1 & 0xFFFFu);                 // tile_dim1 (tile_dim2 = 0)
    g1[5] = (int)(s0 & 0xFFFFFFFFu);              // tensor_dim0_stride[31:0]
    g1[6] = (int)((s0 >> 32) & 0xFFFFu);          // stride0[47:32], dim1_stride[15:0]=0
    g1[7] = 0;                                    // dim1_stride[47:16] = 0
    iv4 g2 = {0, 0, 0, 0};
    iv4 g3 = {0, 0, 0, 0};
    iv8 g4 = {0, 0, 0, 0, 0, 0, 0, 0};
    // 6-arg form (this toolchain): (g0, g1, g2, g3, g4, cpol)
    __builtin_amdgcn_tensor_load_to_lds(g0, g1, g2, g3, g4, 0);
}

// ---------------- kernels ----------------

// h[b,s,:] = wte[id] + wpe[s]
__global__ void embed_kernel(const int* __restrict__ ids,
                             const float* __restrict__ wte,
                             const float* __restrict__ wpe,
                             float* __restrict__ h) {
    int idx = blockIdx.x;          // token index in [0, B*S)
    int e   = threadIdx.x;         // 0..255
    int s   = idx % S_;
    int id  = ids[idx];
    h[(size_t)idx * E_ + e] = wte[(size_t)id * E_ + e] + wpe[(size_t)s * E_ + e];
}

// fp32 [K,N] row-major weight -> bf16 [N,K] row-major (transposed for B-fragments)
__global__ void transpose_to_bf16(const float* __restrict__ in,
                                  u16* __restrict__ out, int K, int N) {
    int i = blockIdx.x * 256 + threadIdx.x;
    if (i >= K * N) return;
    int k = i / N, n = i % N;
    out[(size_t)n * K + k] = f2bf(in[i]);
}

// LayerNorm over E=256; one wave per row, 8 rows per 256-thread block.
__global__ __launch_bounds__(256)
void ln_kernel(const float* __restrict__ x,
               const float* __restrict__ g, const float* __restrict__ bt,
               u16* __restrict__ outB, float* __restrict__ outF) {
    int row  = blockIdx.x * 8 + (threadIdx.x >> 5);
    int lane = threadIdx.x & 31;
    const float* xr = x + (size_t)row * E_;
    float v[8], s = 0.f, s2 = 0.f;
#pragma unroll
    for (int i = 0; i < 8; ++i) {
        v[i] = xr[lane + 32 * i];
        s += v[i];
        s2 += v[i] * v[i];
    }
    s  = rsum32(s);
    s2 = rsum32(s2);
    float mu   = s * (1.0f / E_);
    float var  = s2 * (1.0f / E_) - mu * mu;
    float rstd = rsqrtf(var + 1e-5f);
#pragma unroll
    for (int i = 0; i < 8; ++i) {
        int e = lane + 32 * i;
        float o = (v[i] - mu) * rstd * g[e] + bt[e];
        if (outB) outB[(size_t)row * E_ + e] = f2bf(o);
        else      outF[(size_t)row * E_ + e] = o;
    }
}

// C[M,N] = A_bf16[M,K] x Wt_bf16[N,K]^T + bias.
// Block = 8 waves (2 x 4): 32x128 block tile, 16x32 per wave.
// K staged in 64-wide chunks via Tensor Data Mover into double-buffered LDS;
// wave 0 issues TDM descriptors, TENSORcnt + barrier synchronize the pipeline.
// Epilogue: mode 0 bf16; mode 1 +resid -> f32; mode 2 gelu -> bf16.
#define BK_ 64
__global__ __launch_bounds__(256)
void gemm_wmma(const u16* __restrict__ A, const u16* __restrict__ Wt,
               const float* __restrict__ bias, const float* __restrict__ resid,
               float* __restrict__ outF, u16* __restrict__ outB,
               int M, int N, int K, int mode) {
    __shared__ u16 ldsA[2][32 * BK_];      //  8 KB
    __shared__ u16 ldsB[2][128 * BK_];     // 32 KB

    const int m0   = blockIdx.y * 32;
    const int n0   = blockIdx.x * 128;
    const int wid  = threadIdx.x >> 5;
    const int wm   = wid >> 2;             // 0..1 -> 16-row slice
    const int wn   = wid & 3;              // 0..3 -> 32-col slice
    const int lane = threadIdx.x & 31;
    const int hi   = lane >> 4;
    const int mrow = lane & 15;
    const int coln = lane & 15;
    const bool producer = (wid == 0);

    const unsigned aoff = lds_offset(&ldsA[0][0]);
    const unsigned boff = lds_offset(&ldsB[0][0]);
    const int nk = K / BK_;

    if (producer) {
        tdm_load_2d(A  + (size_t)m0 * K, aoff, BK_,  32, K, K, M);
        tdm_load_2d(Wt + (size_t)n0 * K, boff, BK_, 128, K, K, N);
    }

    v8f c0 = {}, c1 = {};
    for (int kt = 0; kt < nk; ++kt) {
        if (producer) __builtin_amdgcn_s_wait_tensorcnt(0);
        __syncthreads();                               // stage kt visible to all
        if (producer && (kt + 1) < nk) {               // prefetch stage kt+1
            unsigned nb = (kt + 1) & 1;
            tdm_load_2d(A  + (size_t)m0 * K + (kt + 1) * BK_,
                        aoff + nb * 32 * BK_ * 2, BK_, 32, K, K, M);
            tdm_load_2d(Wt + (size_t)n0 * K + (kt + 1) * BK_,
                        boff + nb * 128 * BK_ * 2, BK_, 128, K, K, N);
        }
        const u16* la = &ldsA[kt & 1][0];
        const u16* lb = &ldsB[kt & 1][0];
#pragma unroll
        for (int ks = 0; ks < BK_; ks += 32) {
            Frag a, b0, b1;
            const u16* ar  = la + (wm * 16 + mrow) * BK_ + ks;
            const u16* br0 = lb + (wn * 32 + coln) * BK_ + ks;
            const u16* br1 = lb + (wn * 32 + 16 + coln) * BK_ + ks;
            a.q[0]  = *(const u32x4*)(ar + hi * 16);       // ds_load_b128 pairs
            a.q[1]  = *(const u32x4*)(ar + hi * 16 + 8);
            b0.q[0] = *(const u32x4*)(br0 + hi * 16);
            b0.q[1] = *(const u32x4*)(br0 + hi * 16 + 8);
            b1.q[0] = *(const u32x4*)(br1 + hi * 16);
            b1.q[1] = *(const u32x4*)(br1 + hi * 16 + 8);
            c0 = wmma_bf16(a, b0, c0);
            c1 = wmma_bf16(a, b1, c1);
        }
        __syncthreads();                               // done reading buf kt&1
    }
#pragma unroll
    for (int r = 0; r < 8; ++r) {
        int row  = m0 + wm * 16 + hi * 8 + r;          // C layout: VGPR r <-> row hi*8+r
        int col0 = n0 + wn * 32 + coln, col1 = col0 + 16;
        float v0 = c0[r] + bias[col0];
        float v1 = c1[r] + bias[col1];
        size_t o0 = (size_t)row * N + col0, o1 = (size_t)row * N + col1;
        if (mode == 0)      { outB[o0] = f2bf(v0);            outB[o1] = f2bf(v1); }
        else if (mode == 1) { outF[o0] = v0 + resid[o0];      outF[o1] = v1 + resid[o1]; }
        else                { outB[o0] = f2bf(gelu_new(v0));  outB[o1] = f2bf(gelu_new(v1)); }
    }
}

// Causal flash attention: one wave per (b, head, 16-query tile).
// qkv: bf16 [B*S, 3E]; out: bf16 [B*S, E].
__global__ __launch_bounds__(32)
void attn_kernel(const u16* __restrict__ qkv, u16* __restrict__ out) {
    const int b    = blockIdx.z;
    const int head = blockIdx.y;
    const int q0   = blockIdx.x * 16;
    const int lane = threadIdx.x;
    const int hi   = lane >> 4;
    const int mrow = lane & 15;
    const int coln = lane & 15;

    __shared__ u16 Plds[16 * 32];      // probabilities tile (16 queries x 32 keys)

    Frag qa;
    const u16* qrow = qkv + ((size_t)(b * S_ + q0 + mrow)) * (3 * E_) + head * 32;
    qa.q[0] = *(const u32x4*)(qrow + hi * 16);
    qa.q[1] = *(const u32x4*)(qrow + hi * 16 + 8);

    v8f acc0 = {}, acc1 = {};
    float mst[8], lst[8];
#pragma unroll
    for (int r = 0; r < 8; ++r) { mst[r] = -1e30f; lst[r] = 0.f; }
    const float scale = 0.17677669529663687f;   // 1/sqrt(32)

    for (int k0 = 0; k0 < q0 + 16; k0 += 32) {
        float sc[16];
#pragma unroll
        for (int s = 0; s < 2; ++s) {
            Frag kb;
            const u16* krow = qkv + ((size_t)(b * S_ + k0 + s * 16 + coln)) * (3 * E_)
                              + E_ + head * 32 + hi * 16;
            kb.q[0] = *(const u32x4*)(krow);
            kb.q[1] = *(const u32x4*)(krow + 8);
            v8f zero = {};
            v8f sf = wmma_bf16(qa, kb, zero);
            int key = k0 + s * 16 + coln;
#pragma unroll
            for (int r = 0; r < 8; ++r) {
                int q = q0 + hi * 8 + r;
                float val = sf[r] * scale;
                sc[s * 8 + r] = (key <= q) ? val : -10000.0f;  // masked_fill, as reference
            }
        }
#pragma unroll
        for (int r = 0; r < 8; ++r) {
            float tm = fmaxf(rmax16(sc[r]), rmax16(sc[8 + r]));
            float nm = fmaxf(mst[r], tm);
            float f  = __expf(mst[r] - nm);
            float p0 = __expf(sc[r] - nm);
            float p1 = __expf(sc[8 + r] - nm);
            lst[r] = lst[r] * f + rsum16(p0) + rsum16(p1);
            mst[r] = nm;
            acc0[r] *= f;
            acc1[r] *= f;
            int row = hi * 8 + r;
            Plds[row * 32 + coln]      = f2bf(p0);
            Plds[row * 32 + 16 + coln] = f2bf(p1);
        }
        __syncthreads();
        Frag pa;
        const u16* prow = &Plds[mrow * 32];
        pa.q[0] = *(const u32x4*)(prow + hi * 16);
        pa.q[1] = *(const u32x4*)(prow + hi * 16 + 8);
        Frag vb0, vb1;
#pragma unroll
        for (int j = 0; j < 16; ++j) {
            const u16* vrow = qkv + ((size_t)(b * S_ + k0 + hi * 16 + j)) * (3 * E_)
                              + 2 * E_ + head * 32;
            vb0.u[j] = vrow[coln];
            vb1.u[j] = vrow[16 + coln];
        }
        acc0 = wmma_bf16(pa, vb0, acc0);
        acc1 = wmma_bf16(pa, vb1, acc1);
        __syncthreads();
    }
#pragma unroll
    for (int r = 0; r < 8; ++r) {
        int q = q0 + hi * 8 + r;
        float inv = 1.0f / lst[r];
        u16* orow = out + ((size_t)(b * S_ + q)) * E_ + head * 32;
        orow[coln]      = f2bf(acc0[r] * inv);
        orow[16 + coln] = f2bf(acc1[r] * inv);
    }
}

// ---------------- host orchestration ----------------
extern "C" void kernel_launch(void* const* d_in, const int* in_sizes, int n_in,
                              void* d_out, int out_size, void* d_ws, size_t ws_size,
                              hipStream_t stream) {
    (void)in_sizes; (void)n_in; (void)out_size; (void)ws_size;
    const int*   ids    = (const int*)  d_in[0];
    const float* wte    = (const float*)d_in[1];
    const float* wpe    = (const float*)d_in[2];
    const float* ln1_g  = (const float*)d_in[3];
    const float* ln1_b  = (const float*)d_in[4];
    const float* qkv_w  = (const float*)d_in[5];
    const float* qkv_b  = (const float*)d_in[6];
    const float* proj_w = (const float*)d_in[7];
    const float* proj_b = (const float*)d_in[8];
    const float* ln2_g  = (const float*)d_in[9];
    const float* ln2_b  = (const float*)d_in[10];
    const float* fc1_w  = (const float*)d_in[11];
    const float* fc1_b  = (const float*)d_in[12];
    const float* fc2_w  = (const float*)d_in[13];
    const float* fc2_b  = (const float*)d_in[14];
    const float* lnf_g  = (const float*)d_in[15];
    const float* lnf_b  = (const float*)d_in[16];

    char*  ws  = (char*)d_ws;
    size_t off = 0;
    auto alloc = [&](size_t bytes) -> void* {
        void* p = ws + off;
        off += (bytes + 255) & ~(size_t)255;
        return p;
    };
    float* hbuf    = (float*)alloc((size_t)M_ * E_  * 4);   // residual stream (f32)
    u16*   lnbuf   = (u16*)  alloc((size_t)M_ * E_  * 2);   // LN output (bf16)
    u16*   qkvbuf  = (u16*)  alloc((size_t)M_ * 3 * E_ * 2);
    u16*   attnbuf = (u16*)  alloc((size_t)M_ * E_  * 2);
    u16*   gelubuf = (u16*)  alloc((size_t)M_ * NI_ * 2);
    u16*   wq_t    = (u16*)  alloc((size_t)L_ * 3 * E_ * E_ * 2);
    u16*   wp_t    = (u16*)  alloc((size_t)L_ * E_  * E_ * 2);
    u16*   w1_t    = (u16*)  alloc((size_t)L_ * NI_ * E_ * 2);
    u16*   w2_t    = (u16*)  alloc((size_t)L_ * E_  * NI_ * 2);

    for (int l = 0; l < L_; ++l) {
        transpose_to_bf16<<<(E_ * 3 * E_ + 255) / 256, 256, 0, stream>>>(
            qkv_w + (size_t)l * E_ * 3 * E_, wq_t + (size_t)l * 3 * E_ * E_, E_, 3 * E_);
        transpose_to_bf16<<<(E_ * E_ + 255) / 256, 256, 0, stream>>>(
            proj_w + (size_t)l * E_ * E_, wp_t + (size_t)l * E_ * E_, E_, E_);
        transpose_to_bf16<<<(E_ * NI_ + 255) / 256, 256, 0, stream>>>(
            fc1_w + (size_t)l * E_ * NI_, w1_t + (size_t)l * NI_ * E_, E_, NI_);
        transpose_to_bf16<<<(NI_ * E_ + 255) / 256, 256, 0, stream>>>(
            fc2_w + (size_t)l * NI_ * E_, w2_t + (size_t)l * E_ * NI_, NI_, E_);
    }

    embed_kernel<<<M_, E_, 0, stream>>>(ids, wte, wpe, hbuf);

    for (int l = 0; l < L_; ++l) {
        ln_kernel<<<M_ / 8, 256, 0, stream>>>(hbuf, ln1_g + l * E_, ln1_b + l * E_,
                                              lnbuf, nullptr);
        gemm_wmma<<<dim3(3 * E_ / 128, M_ / 32), 256, 0, stream>>>(
            lnbuf, wq_t + (size_t)l * 3 * E_ * E_, qkv_b + (size_t)l * 3 * E_,
            nullptr, nullptr, qkvbuf, M_, 3 * E_, E_, 0);
        attn_kernel<<<dim3(S_ / 16, H_, B_), 32, 0, stream>>>(qkvbuf, attnbuf);
        gemm_wmma<<<dim3(E_ / 128, M_ / 32), 256, 0, stream>>>(
            attnbuf, wp_t + (size_t)l * E_ * E_, proj_b + (size_t)l * E_,
            hbuf, hbuf, nullptr, M_, E_, E_, 1);
        ln_kernel<<<M_ / 8, 256, 0, stream>>>(hbuf, ln2_g + l * E_, ln2_b + l * E_,
                                              lnbuf, nullptr);
        gemm_wmma<<<dim3(NI_ / 128, M_ / 32), 256, 0, stream>>>(
            lnbuf, w1_t + (size_t)l * NI_ * E_, fc1_b + (size_t)l * NI_,
            nullptr, nullptr, gelubuf, M_, NI_, E_, 2);
        gemm_wmma<<<dim3(E_ / 128, M_ / 32), 256, 0, stream>>>(
            gelubuf, w2_t + (size_t)l * E_ * NI_, fc2_b + (size_t)l * E_,
            hbuf, hbuf, nullptr, M_, E_, NI_, 1);
    }

    ln_kernel<<<M_ / 8, 256, 0, stream>>>(hbuf, lnf_g, lnf_b, nullptr, (float*)d_out);
}
